// HL_HGAT_attpool_87247965651031
// MI455X (gfx1250) — matrix operand order repaired and checked by reading, not compiled, not run on hardware
//
#include <hip/hip_runtime.h>

// ---------------------------------------------------------------------------
// HL-HGAT attention-pool network for MI455X (gfx1250, wave32, WMMA + TDM).
// Dense tall-skinny GEMMs -> v_wmma_f32_16x16x32_f16 (f32 stored, f16 in-reg,
// f32 accumulate; the net is HBM-bandwidth-bound at AI ~ N/2 FLOP/B, so all
// conversions happen in registers / LDS and cost zero extra HBM bytes).
// W k-slices are DMA'd into LDS by the Tensor Data Mover (tensor_load_to_lds
// + s_wait_tensorcnt), then swizzled once into B-fragment order so each lane
// reads its fragment as one contiguous 32B LDS load.
// SpMM / pooling -> gather + global atomic f32.  BN uses batch column stats.
// DenseNet concat handled via column offsets into wide per-level arenas.
// ---------------------------------------------------------------------------

#define LAM_C 0.9f
#define EPS_C 1e-5f
#define SLOPE_C 0.1f

typedef unsigned int u32;
typedef __attribute__((ext_vector_type(16))) _Float16 v16h;
typedef __attribute__((ext_vector_type(8)))  float    v8f;
typedef unsigned int v4u __attribute__((ext_vector_type(4)));
typedef int          v8i __attribute__((ext_vector_type(8)));
typedef int          v4i __attribute__((ext_vector_type(4)));

#if defined(__HIP_DEVICE_COMPILE__) && __has_builtin(__builtin_amdgcn_tensor_load_to_lds)
#define HAVE_TDM 1
#else
#define HAVE_TDM 0
#endif

#if HAVE_TDM
// Issue a TDM load of a [rows<=32] x 16-col f32 tile (row stride N elements)
// from gptr into LDS at byte offset lds_off.  D# layout per CDNA5 ISA §8.3/8.4.
__device__ __forceinline__ void tdm_load_tile(u32 lds_off, const float* gptr,
                                              int rows_avail, int N)
{
  unsigned long long ga = (unsigned long long)(size_t)gptr;
  v4u g0;
  g0[0] = 1u;                                             // count=1, user mode
  g0[1] = lds_off;                                        // lds_addr (bytes)
  g0[2] = (u32)ga;                                        // global_addr[31:0]
  g0[3] = ((u32)(ga >> 32) & 0x01FFFFFFu) | (2u << 30);   // addr[56:32] | type=2
  v8i g1;
  g1[0] = (2 << 16);                   // data_size = 4B; no mask/pad/iterate
  g1[1] = (16 << 16);                  // [63:48] tensor_dim0 = 16 cols
  g1[2] = (rows_avail & 0xFFFF) << 16; // [79:64] dim0 hi = 0 | [95:80] tensor_dim1 lo
  g1[3] = ((rows_avail >> 16) & 0xFFFF) | (16 << 16); // dim1 hi | [127:112] tile_dim0=16
  g1[4] = 32;                          // [143:128] tile_dim1 = 32 rows
  g1[5] = N;                           // tensor_dim0_stride[31:0] (elements)
  g1[6] = 0;
  g1[7] = 0;
  v4i z4 = {0, 0, 0, 0};
#if defined(__clang_major__) && (__clang_major__ >= 23)
  v8i z8 = {0, 0, 0, 0, 0, 0, 0, 0};
  __builtin_amdgcn_tensor_load_to_lds(g0, g1, z4, z4, z8, 0);
#else
  __builtin_amdgcn_tensor_load_to_lds(g0, g1, z4, z4, 0);
#endif
}
#endif

// ---------------------------- kernels --------------------------------------

// Y[M,N] (+)= X[M,K] @ W[K,N]  (+bias).  One wave -> one 16x16 tile.
// 8 waves per 256-thread block; grid.y tiles N (N % 16 == 0).
__global__ __launch_bounds__(256) void k_gemm(
    const float* __restrict__ X, int ldx,
    const float* __restrict__ W,              // row-major [K,N], ld = N
    float* __restrict__ Y, int ldy,
    const float* __restrict__ bias,           // may be null
    int M, int K, int N, int accum)
{
#if HAVE_TDM
  __shared__ __align__(16) float wraw[32 * 16];   // TDM landing pad (f32)
#endif
  __shared__ __align__(32) _Float16 wlds[32 * 16]; // B-fragment-order f16
  const int lane = threadIdx.x & 31;
  const int wave = threadIdx.x >> 5;
  const int l15  = lane & 15;
  const int g    = lane >> 4;                 // lane half (A/B layout group)
  const int tileM   = (blockIdx.x * 8 + wave) * 16;
  const int colBase = blockIdx.y * 16;
  const int row  = tileM + l15;
  const long xrow = (long)row * ldx;

  v8f acc = {};

  for (int k0 = 0; k0 < K; k0 += 32) {
#if HAVE_TDM
    // --- stage W[k0:k0+32, colBase:colBase+16] via Tensor Data Mover ---
    if (wave == 0) {
      int avail = K - k0; if (avail > 32) avail = 32;   // OOB rows zero-filled
      tdm_load_tile((u32)(size_t)(void*)wraw, W + (long)k0 * N + colBase, avail, N);
      __builtin_amdgcn_s_wait_tensorcnt(0);
    }
    __syncthreads();
    {
      // convert + swizzle into B-fragment order: value (kk,cc) -> lane
      // L=((kk>>3)&1)*16+cc, elem e=(kk&7)+((kk>>4)<<3).  Thread handles kk, kk+1 (same L,cc).
      int kk = (threadIdx.x >> 4) * 2;
      int cc = threadIdx.x & 15;
      float w0 = wraw[kk * 16 + cc];
      float w1 = wraw[(kk + 1) * 16 + cc];
      int L = ((kk >> 3) & 1) * 16 + cc, e = (kk & 7) + ((kk >> 4) << 3);
      wlds[L * 16 + e]     = (_Float16)w0;
      wlds[L * 16 + e + 1] = (_Float16)w1;
    }
#else
    {
      int kk = (threadIdx.x >> 4) * 2;
      int cc = threadIdx.x & 15;
      float w0 = (k0 + kk     < K) ? W[(long)(k0 + kk) * N + colBase + cc] : 0.f;
      float w1 = (k0 + kk + 1 < K) ? W[(long)(k0 + kk + 1) * N + colBase + cc] : 0.f;
      int L = ((kk >> 3) & 1) * 16 + cc, e = (kk & 7) + ((kk >> 4) << 3);
      wlds[L * 16 + e]     = (_Float16)w0;
      wlds[L * 16 + e + 1] = (_Float16)w1;
    }
#endif
    __syncthreads();

    // --- A fragment: 16-bit A 16x32 layout; lane half g holds two 8-elem
    //     contiguous runs: K = k0+g*8 .. +7  and  K = k0+16+g*8 .. +7 ---
    v16h a;
    if (k0 + 32 <= K && row < M) {
      const float4* p0 = (const float4*)(X + xrow + k0 + g * 8);
      const float4* p1 = (const float4*)(X + xrow + k0 + 16 + g * 8);
      float4 x0 = p0[0], x1 = p0[1], x2 = p1[0], x3 = p1[1];
      a[0] = (_Float16)x0.x; a[1] = (_Float16)x0.y; a[2] = (_Float16)x0.z; a[3] = (_Float16)x0.w;
      a[4] = (_Float16)x1.x; a[5] = (_Float16)x1.y; a[6] = (_Float16)x1.z; a[7] = (_Float16)x1.w;
      a[8] = (_Float16)x2.x; a[9] = (_Float16)x2.y; a[10] = (_Float16)x2.z; a[11] = (_Float16)x2.w;
      a[12] = (_Float16)x3.x; a[13] = (_Float16)x3.y; a[14] = (_Float16)x3.z; a[15] = (_Float16)x3.w;
      if (k0 + 32 < K) __builtin_prefetch(&X[xrow + k0 + 32], 0, 0);
    } else {
#pragma unroll
      for (int e = 0; e < 16; ++e) {
        int kk = g * 8 + e + ((e >= 8) ? 8 : 0);
        float x = 0.f;
        if (row < M && (k0 + kk) < K) x = X[xrow + k0 + kk];
        a[e] = (_Float16)x;
      }
    }
    // --- B fragment: one contiguous 32B LDS read per lane ---
    v16h b = *(const v16h*)(wlds + lane * 16);

    acc = __builtin_amdgcn_wmma_f32_16x16x32_f16(
        false, a, false, b, (short)0, acc, false, false);
    __syncthreads();
  }

  float bv = bias ? bias[colBase + l15] : 0.f;
#pragma unroll
  for (int r = 0; r < 8; ++r) {
    int rr = tileM + g * 8 + r;               // C/D layout: VGPR r -> M = r (+8 hi half)
    if (rr < M) {
      long o = (long)rr * ldy + colBase + l15;
      float v = acc[r] + bv;
      Y[o] = accum ? (Y[o] + v) : v;
    }
  }
}

__global__ void k_fill2d(float* __restrict__ Y, int ldy, long M, int C, float v) {
  long t = (long)blockIdx.x * blockDim.x + threadIdx.x;
  if (t >= M * (long)C) return;
  long m = t / C; int c = (int)(t - m * C);
  Y[m * (long)ldy + c] = v;
}

// segment_sum SpMM: Y[rows[e], c] += f(val[e]) * X[cols[e], c]
__global__ void k_spmm(const int* __restrict__ rows, const int* __restrict__ cols,
                       const float* __restrict__ val,
                       const float* __restrict__ X, int ldx,
                       float* __restrict__ Y, int ldy,
                       long nnz, int C, int useAbs)
{
  long t = (long)blockIdx.x * blockDim.x + threadIdx.x;
  if (t >= nnz * (long)C) return;
  long e = t / C; int c = (int)(t - e * C);
  float v = val[e];
  if (useAbs) v = fabsf(v);
  int r = rows[e], cl = cols[e];
  atomicAdd(&Y[(long)r * ldy + c], v * X[(long)cl * ldx + c]);
}

// Laguerre terms from x and Ax: T1 = x-Ax; T2 = (-Ax+3T1-x)/2; T3 = (-Ax+5T2-2T1)/3
__global__ void k_laguerre(const float* __restrict__ x, int ldx,
                           const float* __restrict__ Ax,
                           float* __restrict__ T1, float* __restrict__ T2,
                           float* __restrict__ T3, long M, int C)
{
  long t = (long)blockIdx.x * blockDim.x + threadIdx.x;
  if (t >= M * (long)C) return;
  long m = t / C; int c = (int)(t - m * C);
  float xv = x[m * (long)ldx + c], av = Ax[t];
  float t1 = xv - av;
  float t2 = (-av + 3.f * t1 - xv) * 0.5f;
  float t3 = (-av + 5.f * t2 - 2.f * t1) * (1.f / 3.f);
  T1[t] = t1; T2[t] = t2; T3[t] = t3;
}

__global__ void k_scale_rows(float* __restrict__ Y, int ldy, long M, int C,
                             const float* __restrict__ D, float cst)
{
  long t = (long)blockIdx.x * blockDim.x + threadIdx.x;
  if (t >= M * (long)C) return;
  long m = t / C; int c = (int)(t - m * C);
  float s = D ? (1.f / D[m]) : cst;
  Y[m * (long)ldy + c] *= s;
}

// per-column sum / sumsq -> stats[c], stats[N+c]
__global__ __launch_bounds__(256) void k_colstats(const float* __restrict__ Y, int ldy,
                                                  long M, float* __restrict__ stats, int N)
{
  __shared__ float s1[256], s2[256];
  int c = blockIdx.y;
  long m = (long)blockIdx.x * 256 + threadIdx.x;
  float v = (m < M) ? Y[m * (long)ldy + c] : 0.f;
  s1[threadIdx.x] = v; s2[threadIdx.x] = v * v;
  __syncthreads();
  for (int s = 128; s > 0; s >>= 1) {
    if ((int)threadIdx.x < s) { s1[threadIdx.x] += s1[threadIdx.x + s];
                                s2[threadIdx.x] += s2[threadIdx.x + s]; }
    __syncthreads();
  }
  if (threadIdx.x == 0) { atomicAdd(&stats[c], s1[0]); atomicAdd(&stats[N + c], s2[0]); }
}

// y = g*(y-mu)*rsqrt(var+eps)+be, then act: 0 none, 1 leaky(0.1), 2 relu
__global__ void k_bn_act(float* __restrict__ Y, int ldy, long M, int N,
                         const float* __restrict__ stats,
                         const float* __restrict__ gam, const float* __restrict__ bet,
                         int act)
{
  long t = (long)blockIdx.x * blockDim.x + threadIdx.x;
  if (t >= M * (long)N) return;
  long m = t / N; int c = (int)(t - m * N);
  float mu  = stats[c] / (float)M;
  float var = stats[N + c] / (float)M - mu * mu;
  float y = Y[m * (long)ldy + c];
  float o = gam[c] * (y - mu) * rsqrtf(var + EPS_C) + bet[c];
  if (act == 1) o = (o >= 0.f) ? o : SLOPE_C * o;
  else if (act == 2) o = fmaxf(o, 0.f);
  Y[m * (long)ldy + c] = o;
}

// a = sigmoid(((1-lam)*dot(Q1,K) + lam*dot(Q2,K)) / sqrt(32))
__global__ void k_att(const float* __restrict__ Q1, const float* __restrict__ Kv,
                      const float* __restrict__ Q2, long M, float* __restrict__ a)
{
  long m = (long)blockIdx.x * blockDim.x + threadIdx.x;
  if (m >= M) return;
  const float* q1 = Q1 + m * 32; const float* kk = Kv + m * 32; const float* q2 = Q2 + m * 32;
  float s1 = 0.f, s2 = 0.f;
#pragma unroll
  for (int i = 0; i < 32; ++i) { float kv = kk[i]; s1 += q1[i] * kv; s2 += q2[i] * kv; }
  float z = ((1.f - LAM_C) * s1 + LAM_C * s2) * 0.1767766952966369f; // 1/sqrt(32)
  a[m] = 1.f / (1.f + __expf(-z));
}

// Y[pos[m], c] += a[m] * X[m, c]
__global__ void k_pool(const float* __restrict__ X, int ldx, const float* __restrict__ a,
                       const int* __restrict__ pos, long M, int C,
                       float* __restrict__ Y, int ldy)
{
  long t = (long)blockIdx.x * blockDim.x + threadIdx.x;
  if (t >= M * (long)C) return;
  long m = t / C; int c = (int)(t - m * C);
  atomicAdd(&Y[(long)pos[m] * ldy + c], a[m] * X[m * (long)ldx + c]);
}

// r[m] = dot(X[m,:], w) + b
__global__ void k_rowdot(const float* __restrict__ X, int ldx, int C,
                         const float* __restrict__ w, const float* __restrict__ b,
                         long M, float* __restrict__ r)
{
  long m = (long)blockIdx.x * blockDim.x + threadIdx.x;
  if (m >= M) return;
  float s = b[0];
  for (int c = 0; c < C; ++c) s += X[m * (long)ldx + c] * w[c];
  r[m] = s;
}

__global__ void k_set1(float* p, const float* b) { p[0] = b[0]; }

__global__ __launch_bounds__(256) void k_dotacc(const float* __restrict__ x,
                                                const float* __restrict__ w, long n,
                                                float* __restrict__ out)
{
  __shared__ float red[256];
  long i = (long)blockIdx.x * 256 + threadIdx.x;
  red[threadIdx.x] = (i < n) ? x[i] * w[i] : 0.f;
  __syncthreads();
  for (int s = 128; s > 0; s >>= 1) {
    if ((int)threadIdx.x < s) red[threadIdx.x] += red[threadIdx.x + s];
    __syncthreads();
  }
  if (threadIdx.x == 0) atomicAdd(out, red[0]);
}

// ---------------------------- host orchestration ---------------------------

extern "C" void kernel_launch(void* const* d_in, const int* in_sizes, int n_in,
                              void* d_out, int out_size, void* d_ws, size_t ws_size,
                              hipStream_t stream)
{
  (void)in_sizes; (void)n_in; (void)out_size; (void)ws_size;

  const int NLV[3]  = {100000, 50000, 25000};
  const int ELV[3]  = {300000, 150000, 75000};
  const int NNZT[3] = {700000, 350000, 175000};
  const int NNZS[3] = {1500000, 750000, 375000};
  const int FIL[3]  = {32, 64, 128};
  const int FULLW[3] = {96, 224, 480};

  const float* x_t_in = (const float*)d_in[0];
  const float* x_s_in = (const float*)d_in[1];

  struct Lv { const int* ati; const float* atv; long nnzt;
              const int* asi; const float* asv; long nnzs;
              const int* pi;  const float* pv;  long nnzp;
              const float* D; long N, E; };
  Lv lv[3];
  for (int l = 0; l < 3; ++l) {
    int b = 2 + l * 7;
    lv[l].ati = (const int*)d_in[b + 0]; lv[l].atv = (const float*)d_in[b + 1]; lv[l].nnzt = NNZT[l];
    lv[l].asi = (const int*)d_in[b + 2]; lv[l].asv = (const float*)d_in[b + 3]; lv[l].nnzs = NNZS[l];
    lv[l].pi  = (const int*)d_in[b + 4]; lv[l].pv  = (const float*)d_in[b + 5]; lv[l].nnzp = 2L * ELV[l];
    lv[l].D   = (const float*)d_in[b + 6];
    lv[l].N = NLV[l]; lv[l].E = ELV[l];
  }
  const int* pos_t[2] = {(const int*)d_in[23], (const int*)d_in[25]};
  const int* pos_s[2] = {(const int*)d_in[24], (const int*)d_in[26]};

  // params: jax pytree leaf order (dict keys sorted):
  // att[2]{WKe,WKn,WQe,WQn,bKe,bKn,bQe,bQn} -> 27..42
  // blocks[6]{csW,csb,csbe,csg,ctW,ctb,ctbe,ctg,
  //           eW1,eW2,eb1,eb2,ebe1,ebe2,eg1,eg2,
  //           nW1,nW2,nb1,nb2,nbe1,nbe2,ng1,ng2} -> 43..186
  // init_s{W,b,be,g} -> 187..190 ; init_t -> 191..194
  // out{W,b} -> 195..196 ; ro{Ws,Wt,bs,bt} -> 197..200
  auto attP = [&](int i, int j) { return (const float*)d_in[27 + i * 8 + j]; };
  auto blkP = [&](int b, int j) { return (const float*)d_in[43 + b * 24 + j]; };
  const float* initS[4]; for (int j = 0; j < 4; ++j) initS[j] = (const float*)d_in[187 + j];
  const float* initT[4]; for (int j = 0; j < 4; ++j) initT[j] = (const float*)d_in[191 + j];
  const float* Wo  = (const float*)d_in[195];
  const float* bo  = (const float*)d_in[196];
  const float* Ws_ = (const float*)d_in[197];
  const float* Wt_ = (const float*)d_in[198];
  const float* bs_ = (const float*)d_in[199];
  const float* bt_ = (const float*)d_in[200];

  // workspace bump allocator
  size_t off = 0;
  auto alloc = [&](size_t elems) -> float* {
    float* p = (float*)((char*)d_ws + off);
    off += ((elems * sizeof(float) + 255) & ~(size_t)255);
    return p;
  };
  float* XT[3] = { alloc(9600000),  alloc(11200000), alloc(12000000) };  // node arenas
  float* XS[3] = { alloc(28800000), alloc(33600000), alloc(36000000) };  // edge arenas
  float* S2T = alloc(11200000);  // transfer s->t (max N*d)
  float* T2S = alloc(33600000);  // transfer t->s (max E*d)
  float* XN  = alloc(3200000);   // node mlp out [N, dv]
  float* SN  = alloc(9600000);   // edge mlp out [E, dv]
  float* H   = alloc(9600000);   // mlp hidden
  float* AX  = alloc(9600000);   // spmm result / att lin 1
  float* T1B = alloc(9600000);   //                 att lin 2
  float* T2B = alloc(9600000);   //                 att lin 3
  float* T3B = alloc(9600000);
  float* ATt = alloc(300000);
  float* ATs = alloc(300000);
  float* STATS = alloc(1024);
  float* RT = alloc(25000);
  float* RS = alloc(75000);

  auto cdiv = [](long a, long b) { return (unsigned)((a + b - 1) / b); };

  auto fill = [&](float* Y, int ldy, long M, int C, float v) {
    k_fill2d<<<cdiv(M * (long)C, 256), 256, 0, stream>>>(Y, ldy, M, C, v);
  };
  auto gemm = [&](const float* X, int ldx, const float* W, float* Y, int ldy,
                  const float* bias, long M, int K, int N, int accum) {
    dim3 grid(cdiv(M, 128), (unsigned)(N / 16));
    k_gemm<<<grid, 256, 0, stream>>>(X, ldx, W, Y, ldy, bias, (int)M, K, N, accum);
  };
  auto bn = [&](float* Y, int ldy, long M, int N, const float* g_, const float* be_, int act) {
    fill(STATS, 1, 2 * N, 1, 0.f);
    dim3 grid(cdiv(M, 256), (unsigned)N);
    k_colstats<<<grid, 256, 0, stream>>>(Y, ldy, M, STATS, N);
    k_bn_act<<<cdiv(M * (long)N, 256), 256, 0, stream>>>(Y, ldy, M, N, STATS, g_, be_, act);
  };
  auto spmm = [&](const int* rows, const int* cols, const float* val, long nnz,
                  const float* X, int ldx, float* Y, int ldy, long Mout, int C, int useAbs) {
    fill(Y, ldy, Mout, C, 0.f);
    k_spmm<<<cdiv(nnz * (long)C, 256), 256, 0, stream>>>(rows, cols, val, X, ldx, Y, ldy, nnz, C, useAbs);
  };
  auto hlconv = [&](const float* x, int ldx, long M, int Cin, int Cout,
                    const int* aidx, const float* aval, long nnz,
                    const float* W, const float* b, const float* g_, const float* be_,
                    float* Y, int ldy) {
    spmm(aidx, aidx + nnz, aval, nnz, x, ldx, AX, Cin, M, Cin, 0);
    k_laguerre<<<cdiv(M * (long)Cin, 256), 256, 0, stream>>>(x, ldx, AX, T1B, T2B, T3B, M, Cin);
    gemm(x,   ldx, W,                          Y, ldy, b,       M, Cin, Cout, 0);
    gemm(T1B, Cin, W + (long)1 * Cin * Cout,   Y, ldy, nullptr, M, Cin, Cout, 1);
    gemm(T2B, Cin, W + (long)2 * Cin * Cout,   Y, ldy, nullptr, M, Cin, Cout, 1);
    gemm(T3B, Cin, W + (long)3 * Cin * Cout,   Y, ldy, nullptr, M, Cin, Cout, 1);
    bn(Y, ldy, M, Cout, g_, be_, 1);
  };
  auto transfer = [&](const float* xt, int ldt, long Mt, const float* xs, int lds_, long Ms,
                      const int* pidx, const float* pval, long nnzp, const float* D, int C) {
    spmm(pidx, pidx + nnzp, pval, nnzp, xs, lds_, S2T, C, Mt, C, 1);
    k_scale_rows<<<cdiv(Mt * (long)C, 256), 256, 0, stream>>>(S2T, C, Mt, C, D, 1.f);
    spmm(pidx + nnzp, pidx, pval, nnzp, xt, ldt, T2S, C, Ms, C, 1);
    k_scale_rows<<<cdiv(Ms * (long)C, 256), 256, 0, stream>>>(T2S, C, Ms, C, nullptr, 0.5f);
  };
  auto mlp = [&](const float* z1, int ld1, const float* z2, int ld2, long M, int d, int dv,
                 const float* W1, const float* b1, const float* g1, const float* be1,
                 const float* W2, const float* b2, const float* g2, const float* be2,
                 float* outb) {
    gemm(z1, ld1, W1,                 H, dv, b1,      M, d,  dv, 0);
    gemm(z2, ld2, W1 + (long)d * dv,  H, dv, nullptr, M, d,  dv, 1);  // concat = split-K
    bn(H, dv, M, dv, g1, be1, 2);
    gemm(H, dv, W2, outb, dv, b2, M, dv, dv, 0);
    bn(outb, dv, M, dv, g2, be2, 2);
  };

  // ---- initial Hodge-Laguerre convs ----
  hlconv(x_t_in, 64, NLV[0], 64, 32, lv[0].ati, lv[0].atv, lv[0].nnzt,
         initT[0], initT[1], initT[3], initT[2], XT[0], 96);
  hlconv(x_s_in, 1,  ELV[0], 1,  32, lv[0].asi, lv[0].asv, lv[0].nnzs,
         initS[0], initS[1], initS[3], initS[2], XS[0], 96);

  float* xt = XT[0]; int ldt = 96;
  float* xs = XS[0]; int lds_ = 96;
  int wt = 32, bi = 0;

  for (int i = 0; i < 3; ++i) {
    long Mt = lv[i].N, Ms = lv[i].E;
    for (int ch = 0; ch < 2; ++ch, ++bi) {
      int d = wt, dv = FIL[i];
      transfer(xt, ldt, Mt, xs, lds_, Ms, lv[i].pi, lv[i].pv, lv[i].nnzp, lv[i].D, d);
      mlp(S2T, d, xt, ldt, Mt, d, dv,
          blkP(bi, 16), blkP(bi, 18), blkP(bi, 22), blkP(bi, 20),   // nW1,nb1,ng1,nbe1
          blkP(bi, 17), blkP(bi, 19), blkP(bi, 23), blkP(bi, 21),   // nW2,nb2,ng2,nbe2
          XN);
      mlp(T2S, d, xs, lds_, Ms, d, dv,
          blkP(bi, 8),  blkP(bi, 10), blkP(bi, 14), blkP(bi, 12),   // eW1,eb1,eg1,ebe1
          blkP(bi, 9),  blkP(bi, 11), blkP(bi, 15), blkP(bi, 13),   // eW2,eb2,eg2,ebe2
          SN);
      hlconv(XN, dv, Mt, dv, dv, lv[i].ati, lv[i].atv, lv[i].nnzt,
             blkP(bi, 4), blkP(bi, 5), blkP(bi, 7), blkP(bi, 6),    // ctW,ctb,ctg,ctbe
             xt + wt, ldt);                                          // DenseNet concat in-place
      hlconv(SN, dv, Ms, dv, dv, lv[i].asi, lv[i].asv, lv[i].nnzs,
             blkP(bi, 0), blkP(bi, 1), blkP(bi, 3), blkP(bi, 2),    // csW,csb,csg,csbe
             xs + wt, lds_);
      wt += dv;
    }
    if (i < 2) {
      int d = wt;
      transfer(xt, ldt, Mt, xs, lds_, Ms, lv[i].pi, lv[i].pv, lv[i].nnzp, lv[i].D, d);
      // a_t = sigmoid(((1-lam)<Qe(s2t),Kn(t)> + lam<Qn(t),Kn(t)>)/sqrt(dk))
      gemm(S2T, d,   attP(i, 2), AX,  32, attP(i, 6), Mt, d, 32, 0); // WQe,bQe
      gemm(xt,  ldt, attP(i, 1), T1B, 32, attP(i, 5), Mt, d, 32, 0); // WKn,bKn
      gemm(xt,  ldt, attP(i, 3), T2B, 32, attP(i, 7), Mt, d, 32, 0); // WQn,bQn
      k_att<<<cdiv(Mt, 256), 256, 0, stream>>>(AX, T1B, T2B, Mt, ATt);
      // a_s = sigmoid(((1-lam)<Qn(t2s),Ke(s)> + lam<Qe(s),Ke(s)>)/sqrt(dk))
      gemm(T2S, d,    attP(i, 3), AX,  32, attP(i, 7), Ms, d, 32, 0); // WQn,bQn
      gemm(xs,  lds_, attP(i, 0), T1B, 32, attP(i, 4), Ms, d, 32, 0); // WKe,bKe
      gemm(xs,  lds_, attP(i, 2), T2B, 32, attP(i, 6), Ms, d, 32, 0); // WQe,bQe
      k_att<<<cdiv(Ms, 256), 256, 0, stream>>>(AX, T1B, T2B, Ms, ATs);
      // attention pooling into next level arenas (cols 0..wt-1)
      long Mt2 = lv[i + 1].N, Ms2 = lv[i + 1].E;
      fill(XT[i + 1], FULLW[i + 1], Mt2, wt, 0.f);
      k_pool<<<cdiv(Mt * (long)wt, 256), 256, 0, stream>>>(xt, ldt, ATt, pos_t[i], Mt, wt,
                                                           XT[i + 1], FULLW[i + 1]);
      fill(XS[i + 1], FULLW[i + 1], Ms2, wt, 0.f);
      k_pool<<<cdiv(Ms * (long)wt, 256), 256, 0, stream>>>(xs, lds_, ATs, pos_s[i], Ms, wt,
                                                           XS[i + 1], FULLW[i + 1]);
      xt = XT[i + 1]; ldt = FULLW[i + 1];
      xs = XS[i + 1]; lds_ = FULLW[i + 1];
    }
  }

  // ---- readout: last block outputs live at column 352 of level-2 arenas ----
  k_rowdot<<<cdiv(25000, 256), 256, 0, stream>>>(XT[2] + 352, 480, 128, Wt_, bt_, 25000, RT);
  k_rowdot<<<cdiv(75000, 256), 256, 0, stream>>>(XS[2] + 352, 480, 128, Ws_, bs_, 75000, RS);
  k_set1<<<1, 1, 0, stream>>>((float*)d_out, bo);
  k_dotacc<<<cdiv(25000, 256), 256, 0, stream>>>(RT, Wo,         25000, (float*)d_out);
  k_dotacc<<<cdiv(75000, 256), 256, 0, stream>>>(RS, Wo + 25000, 75000, (float*)d_out);
}
